// RNNDecoder_12773232738961
// MI455X (gfx1250) — compile-verified, matrix operand order checked by reading, not run on hardware
//
#include <hip/hip_runtime.h>
#include <hip/hip_bf16.h>

typedef unsigned short u16;
typedef __attribute__((ext_vector_type(16))) __bf16 v16bf;
typedef __attribute__((ext_vector_type(8)))  __bf16 v8bf;
typedef __attribute__((ext_vector_type(8)))  float  v8f;

#define BB 32
#define TT 64
#define SS 128
#define DD 512
#define VV 32000
#define NL 6
#define G3 1536   // 3*D
#define GIN 1024  // D + D

// ---------- helpers ----------
__device__ __forceinline__ u16 f2bf(float x) {
    unsigned int u = __float_as_uint(x);
    unsigned int r = u + 0x7FFFu + ((u >> 16) & 1u);   // round-to-nearest-even
    return (u16)(r >> 16);
}

// Load a 16x32 bf16 WMMA A-fragment (or K x 16 B-fragment from an
// out_features-major weight matrix) per the CDNA5 ISA 16-bit A layout:
// lane (m = lane&15, hi = lane>>4); elements 0..7 -> k0+hi*8+e,
// elements 8..15 -> k0+16+hi*8+(e-8). Two 16B contiguous loads per lane.
// Works for both global and LDS pointers (addrspace inferred after inlining
// -> ds_load_b128 for __shared__ sources).
__device__ __forceinline__ v16bf load_frag(const u16* X, int row0, int ldx, int k0, int lane) {
    int m  = lane & 15;
    int hb = lane >> 4;
    const u16* p = X + (size_t)(row0 + m) * ldx + k0 + hb * 8;
    union { v16bf v; struct { v8bf lo; v8bf hi; } s; } u;
    u.s.lo = *(const v8bf*)(p);
    u.s.hi = *(const v8bf*)(p + 16);
    return u.v;
}

#define WMMA_BF16(A, B, C) \
    __builtin_amdgcn_wmma_f32_16x16x32_bf16(false, (A), false, (B), (short)0, (C), false, false)

// ---------- one-time conversion kernels ----------
__global__ void k_cvt(const float* __restrict__ s, u16* __restrict__ d, int n) {
    for (int i = blockIdx.x * blockDim.x + threadIdx.x; i < n; i += gridDim.x * blockDim.x)
        d[i] = f2bf(s[i]);
}
__global__ void k_cvt_add(const float* __restrict__ a, const float* __restrict__ b,
                          u16* __restrict__ d, int n) {
    for (int i = blockIdx.x * blockDim.x + threadIdx.x; i < n; i += gridDim.x * blockDim.x)
        d[i] = f2bf(a[i] + b[i]);
}
__global__ void k_bias3(const float* __restrict__ a, const float* __restrict__ b,
                        const float* __restrict__ c, float* __restrict__ d, int n) {
    int i = blockIdx.x * blockDim.x + threadIdx.x;
    if (i < n) d[i] = a[i] + b[i] + c[i];
}
// h0 = mean_S enc ; tok = tgt[:,0]
__global__ void k_init(const float* __restrict__ enc, const int* __restrict__ tgt,
                       float* __restrict__ h, int* __restrict__ tok) {
    int idx = blockIdx.x * blockDim.x + threadIdx.x;
    if (idx < BB * DD) {
        int b = idx >> 9, d = idx & (DD - 1);
        float acc = 0.f;
        for (int s = 0; s < SS; ++s) acc += enc[((size_t)b * SS + s) * DD + d];
        h[idx] = acc * (1.0f / SS);
    }
    if (idx < BB) tok[idx] = tgt[idx * TT];
}

// ---------- attention (fp32 VALU; tiny: 4.2M MAC) ----------
__global__ void k_attn(const float* __restrict__ enc, const int* __restrict__ src,
                       const float* __restrict__ emb, const int* __restrict__ tok,
                       const float* __restrict__ h,
                       u16* __restrict__ rnn_bf, u16* __restrict__ h_bf, int layer0) {
    int b = blockIdx.x, tid = threadIdx.x;
    __shared__ float sh[DD];
    __shared__ float ssc[SS];
    __shared__ float sinv;
    for (int d = tid; d < DD; d += 256) sh[d] = h[b * DD + d];
    __syncthreads();
    if (tid < SS) {
        const float* e = enc + ((size_t)b * SS + tid) * DD;
        float acc = 0.f;
        #pragma unroll 4
        for (int d = 0; d < DD; d += 4) {
            float4 ev = *(const float4*)(e + d);
            float4 hv = *(const float4*)(sh + d);
            acc += ev.x * hv.x + ev.y * hv.y + ev.z * hv.z + ev.w * hv.w;
        }
        acc *= 0.044194173824159216f;              // 1/sqrt(512)
        if (src[b * SS + tid] == 0) acc = -__builtin_inff();
        ssc[tid] = acc;
    }
    __syncthreads();
    if (tid == 0) {
        float m = -__builtin_inff();
        for (int s = 0; s < SS; ++s) m = fmaxf(m, ssc[s]);
        float sum = 0.f;
        for (int s = 0; s < SS; ++s) { float e = __expf(ssc[s] - m); ssc[s] = e; sum += e; }
        sinv = 1.0f / sum;
    }
    __syncthreads();
    float inv = sinv;
    for (int d = tid; d < DD; d += 256) {
        float acc = 0.f;
        for (int s = 0; s < SS; ++s) acc += ssc[s] * enc[((size_t)b * SS + s) * DD + d];
        acc *= inv;
        rnn_bf[b * GIN + DD + d] = f2bf(acc);      // context half of rnn_in
    }
    const float* xsrc = layer0 ? (emb + (size_t)tok[b] * DD) : (h + b * DD);
    for (int d = tid; d < DD; d += 256) {
        rnn_bf[b * GIN + d] = f2bf(xsrc[d]);       // x half of rnn_in
        h_bf[b * DD + d]    = f2bf(sh[d]);         // current h in bf16
    }
}

// ---------- GRU GEMMs: gi = rnn_in @ Wih^T (K=1024), gh = h @ Whh^T (K=512) ----------
// 96 n-tiles, one per wave; each wave does 2 M-tiles => 96 WMMAs/wave.
__global__ void k_gru_gemm(const u16* __restrict__ rnn_bf, const u16* __restrict__ h_bf,
                           const u16* __restrict__ Wih, const u16* __restrict__ Whh,
                           float* __restrict__ gi, float* __restrict__ gh) {
    int wave = (blockIdx.x * blockDim.x + threadIdx.x) >> 5;   // 0..95 exact
    int lane = threadIdx.x & 31;
    int n0 = wave * 16;
    v8f ai0 = {}, ai1 = {}, ah0 = {}, ah1 = {};
    for (int k = 0; k < GIN; k += 32) {
        v16bf bf = load_frag(Wih, n0, GIN, k, lane);
        v16bf a0 = load_frag(rnn_bf, 0, GIN, k, lane);
        v16bf a1 = load_frag(rnn_bf, 16, GIN, k, lane);
        ai0 = WMMA_BF16(a0, bf, ai0);
        ai1 = WMMA_BF16(a1, bf, ai1);
    }
    for (int k = 0; k < DD; k += 32) {
        v16bf bf = load_frag(Whh, n0, DD, k, lane);
        v16bf a0 = load_frag(h_bf, 0, DD, k, lane);
        v16bf a1 = load_frag(h_bf, 16, DD, k, lane);
        ah0 = WMMA_BF16(a0, bf, ah0);
        ah1 = WMMA_BF16(a1, bf, ah1);
    }
    int n = n0 + (lane & 15);
    int rb = (lane >> 4) * 8;
    #pragma unroll
    for (int r = 0; r < 8; ++r) {
        gi[(size_t)(rb + r) * G3 + n]        = ai0[r];
        gi[(size_t)(16 + rb + r) * G3 + n]   = ai1[r];
        gh[(size_t)(rb + r) * G3 + n]        = ah0[r];
        gh[(size_t)(16 + rb + r) * G3 + n]   = ah1[r];
    }
}

// ---------- GRU gate math ----------
__global__ void k_gate(const float* __restrict__ gi, const float* __restrict__ gh,
                       const float* __restrict__ bi, const float* __restrict__ bh,
                       float* __restrict__ h, u16* __restrict__ h_bf) {
    int idx = blockIdx.x * blockDim.x + threadIdx.x;
    if (idx >= BB * DD) return;
    int b = idx >> 9, f = idx & (DD - 1);
    const float* gib = gi + (size_t)b * G3;
    const float* ghb = gh + (size_t)b * G3;
    float ir = gib[f]          + bi[f];
    float iz = gib[DD + f]     + bi[DD + f];
    float in = gib[2 * DD + f] + bi[2 * DD + f];
    float hr = ghb[f]          + bh[f];
    float hz = ghb[DD + f]     + bh[DD + f];
    float hn = ghb[2 * DD + f] + bh[2 * DD + f];
    float r = 1.0f / (1.0f + __expf(-(ir + hr)));
    float z = 1.0f / (1.0f + __expf(-(iz + hz)));
    float n = tanhf(in + r * hn);
    float hnew = (1.0f - z) * n + z * h[idx];
    h[idx] = hnew;
    h_bf[idx] = f2bf(hnew);
}

// ---------- vocab projection: out = h@(U+V)^T + ctx@C^T + bias ----------
// 125 WGs x 8 waves, 2 N-tiles per wave (2000 tiles). h/ctx staged in LDS so
// A-fragments come from the DS pipe while B (weight) fragments stream on the
// vmem pipe. 8 WMMAs per wave per k-step with only 4 vmem B-frag loads.
__global__ void k_outproj(const u16* __restrict__ h_bf, const u16* __restrict__ rnn_bf,
                          const u16* __restrict__ UV, const u16* __restrict__ Cw,
                          const float* __restrict__ bias, float* __restrict__ out, int t) {
    __shared__ u16 sh[BB * DD];   // h, 32 KB
    __shared__ u16 sc[BB * DD];   // ctx, 32 KB
    int tid = threadIdx.x;
    // cooperative stage: 16B vector copies; ctx rows are strided in rnn_bf
    for (int i = tid; i < BB * DD / 8; i += 256) {
        int off = i * 8;
        *(v8bf*)(sh + off) = *(const v8bf*)(h_bf + off);
        int b = off >> 9, c = off & (DD - 1);
        *(v8bf*)(sc + off) = *(const v8bf*)(rnn_bf + b * GIN + DD + c);
    }
    __syncthreads();
    int lane = tid & 31;
    int n0 = (blockIdx.x * 8 + (tid >> 5)) * 32;   // two tiles: n0, n0+16
    v8f a00 = {}, a10 = {}, a01 = {}, a11 = {};    // acc[mtile][ntile]
    for (int k = 0; k < DD; k += 32) {
        v16bf ha0 = load_frag(sh, 0, DD, k, lane);     // ds pipe
        v16bf ha1 = load_frag(sh, 16, DD, k, lane);
        v16bf ca0 = load_frag(sc, 0, DD, k, lane);
        v16bf ca1 = load_frag(sc, 16, DD, k, lane);
        v16bf bu0 = load_frag(UV, n0,      DD, k, lane);   // vmem pipe
        v16bf bu1 = load_frag(UV, n0 + 16, DD, k, lane);
        v16bf bc0 = load_frag(Cw, n0,      DD, k, lane);
        v16bf bc1 = load_frag(Cw, n0 + 16, DD, k, lane);
        a00 = WMMA_BF16(ha0, bu0, a00);  a00 = WMMA_BF16(ca0, bc0, a00);
        a10 = WMMA_BF16(ha1, bu0, a10);  a10 = WMMA_BF16(ca1, bc0, a10);
        a01 = WMMA_BF16(ha0, bu1, a01);  a01 = WMMA_BF16(ca0, bc1, a01);
        a11 = WMMA_BF16(ha1, bu1, a11);  a11 = WMMA_BF16(ca1, bc1, a11);
    }
    int nlo = n0 + (lane & 15);
    int nhi = nlo + 16;
    int rb  = (lane >> 4) * 8;
    float bv0 = bias[nlo], bv1 = bias[nhi];
    #pragma unroll
    for (int r = 0; r < 8; ++r) {
        out[((size_t)(rb + r) * TT + t) * VV + nlo]      = a00[r] + bv0;
        out[((size_t)(16 + rb + r) * TT + t) * VV + nlo] = a10[r] + bv0;
        out[((size_t)(rb + r) * TT + t) * VV + nhi]      = a01[r] + bv1;
        out[((size_t)(16 + rb + r) * TT + t) * VV + nhi] = a11[r] + bv1;
    }
}

// ---------- argmax -> next token ----------
__global__ void k_argmax(const float* __restrict__ out, int t, int* __restrict__ tok) {
    int b = blockIdx.x, tid = threadIdx.x;
    const float* row = out + ((size_t)b * TT + t) * VV;
    float best = -__builtin_inff(); int bidx = 0;
    for (int v = tid; v < VV; v += 256) {
        float x = row[v];
        if (x > best) { best = x; bidx = v; }
    }
    __shared__ float sv[256];
    __shared__ int   si[256];
    sv[tid] = best; si[tid] = bidx;
    __syncthreads();
    for (int s = 128; s > 0; s >>= 1) {
        if (tid < s) {
            if (sv[tid + s] > sv[tid] || (sv[tid + s] == sv[tid] && si[tid + s] < si[tid])) {
                sv[tid] = sv[tid + s]; si[tid] = si[tid + s];
            }
        }
        __syncthreads();
    }
    if (tid == 0) tok[b] = si[0];
}

extern "C" void kernel_launch(void* const* d_in, const int* in_sizes, int n_in,
                              void* d_out, int out_size, void* d_ws, size_t ws_size,
                              hipStream_t stream) {
    (void)in_sizes; (void)n_in; (void)out_size; (void)ws_size;
    const int*   tgt  = (const int*)d_in[0];
    const int*   src  = (const int*)d_in[1];
    const float* enc  = (const float*)d_in[2];
    const float* emb  = (const float*)d_in[3];
    const float* W_ih = (const float*)d_in[4];
    const float* W_hh = (const float*)d_in[5];
    const float* b_ih = (const float*)d_in[6];
    const float* b_hh = (const float*)d_in[7];
    const float* U_w  = (const float*)d_in[8];
    const float* U_b  = (const float*)d_in[9];
    const float* V_w  = (const float*)d_in[10];
    const float* V_b  = (const float*)d_in[11];
    const float* C_w  = (const float*)d_in[12];
    const float* C_b  = (const float*)d_in[13];
    float* out = (float*)d_out;

    // workspace carve-up (~94 MB; bf16 weights fit in 192 MB L2)
    char* p = (char*)d_ws;
    auto alloc = [&](size_t bytes) { char* r = p; p += (bytes + 255) & ~(size_t)255; return r; };
    u16*   Wih_bf = (u16*)  alloc((size_t)NL * G3 * GIN * 2);
    u16*   Whh_bf = (u16*)  alloc((size_t)NL * G3 * DD * 2);
    u16*   UV_bf  = (u16*)  alloc((size_t)VV * DD * 2);
    u16*   C_bf   = (u16*)  alloc((size_t)VV * DD * 2);
    float* bias   = (float*)alloc((size_t)VV * 4);
    float* h      = (float*)alloc((size_t)BB * DD * 4);
    u16*   h_bf   = (u16*)  alloc((size_t)BB * DD * 2);
    u16*   rnn_bf = (u16*)  alloc((size_t)BB * GIN * 2);
    float* gi     = (float*)alloc((size_t)BB * G3 * 4);
    float* gh     = (float*)alloc((size_t)BB * G3 * 4);
    int*   tok    = (int*)  alloc((size_t)BB * 4);

    // one-time per launch: precision conversion + folding
    k_cvt    <<<4096, 256, 0, stream>>>(W_ih, Wih_bf, NL * G3 * GIN);
    k_cvt    <<<2048, 256, 0, stream>>>(W_hh, Whh_bf, NL * G3 * DD);
    k_cvt_add<<<8192, 256, 0, stream>>>(U_w, V_w, UV_bf, VV * DD);   // x==h after last layer => fold U+V
    k_cvt    <<<8192, 256, 0, stream>>>(C_w, C_bf, VV * DD);
    k_bias3  <<<125,  256, 0, stream>>>(U_b, V_b, C_b, bias, VV);
    k_init   <<<64,   256, 0, stream>>>(enc, tgt, h, tok);

    for (int t = 0; t < TT; ++t) {
        for (int l = 0; l < NL; ++l) {
            k_attn    <<<BB, 256, 0, stream>>>(enc, src, emb, tok, h, rnn_bf, h_bf, l == 0);
            k_gru_gemm<<<12, 256, 0, stream>>>(rnn_bf, h_bf,
                                               Wih_bf + (size_t)l * G3 * GIN,
                                               Whh_bf + (size_t)l * G3 * DD, gi, gh);
            k_gate    <<<64, 256, 0, stream>>>(gi, gh, b_ih + l * G3, b_hh + l * G3, h, h_bf);
        }
        k_outproj<<<125, 256, 0, stream>>>(h_bf, rnn_bf, UV_bf, C_bf, bias, out, t);
        k_argmax <<<BB,  256, 0, stream>>>(out, t, tok);
    }
}
